// LSTM_88596585382347
// MI455X (gfx1250) — matrix operand order repaired and checked by reading, not compiled, not run on hardware
//
#include <hip/hip_runtime.h>
#include <hip/hip_bf16.h>

// Problem constants (from reference): B=256, S=512, C=64, H=128, 4H=512
#define LB   256
#define LS   512
#define LC   64
#define LH   128
#define LG   512          // 4H
#define NTILES (LG / 16)  // 32 column tiles of the gate matrix

typedef __attribute__((ext_vector_type(16))) __bf16 v16bf;
typedef __attribute__((ext_vector_type(8)))  __bf16 v8bf;
typedef __attribute__((ext_vector_type(8)))  float  v8f;

__device__ __forceinline__ float sig_f(float x) { return 1.0f / (1.0f + __expf(-x)); }
__device__ __forceinline__ float tanh_f(float x) { return 2.0f / (1.0f + __expf(-2.0f * x)) - 1.0f; }

// xg is stored tile-major in the native WMMA C-fragment layout:
//   XG[(rowtile*32 + ntile)*32 + lane][0..7]  (v8f per lane)
__device__ __forceinline__ size_t xg_off(int rowtile, int ntile, int lane) {
  return ((((size_t)rowtile * NTILES) + ntile) * 32 + lane) * 8;
}

// ---------------------------------------------------------------------------
// Kernel 1: pack weight matrix W (512, K) f32 into bf16 B-fragment images:
//   WP[(ntile*(K/32) + kt)*32 + lane][e0..15],  element e -> K = kt*32 + (e&7) + (e>>3)*16 + hi*8
// ---------------------------------------------------------------------------
__global__ __launch_bounds__(256) void pack_w_kernel(const float* __restrict__ W,
                                                     __bf16* __restrict__ WP,
                                                     int K) {
  const int nfrag = NTILES * (K >> 5);
  int idx = blockIdx.x * 256 + threadIdx.x;        // frag*32 + lane
  if (idx >= nfrag * 32) return;
  const int lane  = idx & 31;
  const int frag  = idx >> 5;
  const int nkt   = K >> 5;
  const int kt    = frag % nkt;
  const int ntile = frag / nkt;
  const int hi    = lane >> 4;
  const int n     = ntile * 16 + (lane & 15);
  v16bf t;
#pragma unroll
  for (int e = 0; e < 16; ++e)
    t[e] = (__bf16)W[(size_t)n * K + kt * 32 + (e & 7) + ((e >> 3) << 4) + (hi << 3)];
  *(v16bf*)(WP + (size_t)idx * 16) = t;
}

// ---------------------------------------------------------------------------
// Kernel 2: transpose + convert x (B,S,C) f32 directly into packed bf16
// A-fragment images for layer 0 (K=64):
//   A0P[((rowtile*2 + kt)*32 + lane)*16 + half*8 + 0..7]
//   chunk covers k = kt*32 + half*16 + hi*8 .. +7 of row (rowtile*16 + lm)
// ---------------------------------------------------------------------------
__global__ __launch_bounds__(256) void cvt_x_pack_kernel(const float* __restrict__ x,
                                                         __bf16* __restrict__ a0p) {
  int idx = blockIdx.x * 256 + threadIdx.x;   // total = 8192*2*32*2 = 1048576 chunks
  const int ehalf   = idx & 1;
  const int lane    = (idx >> 1) & 31;
  const int kt      = (idx >> 6) & 1;
  const int rowtile = idx >> 7;
  const int hi = lane >> 4, lm = lane & 15;
  const int row = rowtile * 16 + lm;          // s*256 + b
  const int s = row >> 8, b = row & 255;
  const int c0 = kt * 32 + ehalf * 16 + hi * 8;
  const float* src = x + ((size_t)b * LS + s) * LC + c0;
  v8bf t;
#pragma unroll
  for (int e = 0; e < 8; ++e) t[e] = (__bf16)src[e];
  *(v8bf*)(a0p + (size_t)idx * 8) = t;
}

// ---------------------------------------------------------------------------
// Kernel 3: batched input GEMM (fully pre-packed operands):
//   g[row][n] = sum_k A[row][k]*W[n][k] + bih[n] + bhh[n]
// A: packed bf16 A-fragments [rowtile][kt][lane][16]
// WP: packed bf16 B-fragments [ntile][kt][lane][16]
// Output: tile-major f32 C-fragments (feeds recurrent kernel directly).
// Block = 16 rows x 512 cols, 8 waves; wave w -> cols [64w, 64w+64).
// ---------------------------------------------------------------------------
template <int K>
__global__ __launch_bounds__(256) void ingemm_kernel(const __bf16* __restrict__ A,
                                                     const __bf16* __restrict__ WP,
                                                     const float* __restrict__ bih,
                                                     const float* __restrict__ bhh,
                                                     float* __restrict__ xg) {
  constexpr int NKT = K >> 5;
  const int lane = threadIdx.x & 31;
  const int wave = threadIdx.x >> 5;      // 0..7
  const int lm   = lane & 15;
  const int rowtile = blockIdx.x;

  // A fragments: same for all waves of the block (broadcast from cache)
  v16bf af[NKT];
#pragma unroll
  for (int kt = 0; kt < NKT; ++kt)
    af[kt] = *(const v16bf*)(A + (((size_t)rowtile * NKT + kt) * 32 + lane) * 16);

#pragma unroll
  for (int nt = 0; nt < 4; ++nt) {
    const int ntile = wave * 4 + nt;
    const float bsum = bih[ntile * 16 + lm] + bhh[ntile * 16 + lm];
    v8f acc;
#pragma unroll
    for (int r = 0; r < 8; ++r) acc[r] = bsum;
#pragma unroll
    for (int kt = 0; kt < NKT; ++kt) {
      v16bf bw = *(const v16bf*)(WP + (((size_t)ntile * NKT + kt) * 32 + lane) * 16);
      acc = __builtin_amdgcn_wmma_f32_16x16x32_bf16(false, af[kt], false, bw,
                                                    (short)0, acc, false, false);
    }
    *(v8f*)(xg + xg_off(rowtile, ntile, lane)) = acc;   // one 32B store per lane
  }
}

// ---------------------------------------------------------------------------
// Kernel 4: persistent recurrent LSTM layer.
// Grid = 16 blocks; block owns batch rows [16*blk, 16*blk+16) for all S steps.
// Whh fragments (bf16) live in VGPRs for all 512 steps; per step:
//   g = xg[s] + h @ Whh^T  (WMMA, f32 accum) -> LDS -> fused gates -> h,c
// xg fragments for step s+1 are software-pipelined (loaded during gate phase).
// y output is written directly in layer-1's packed A-fragment layout.
// ---------------------------------------------------------------------------
#define GST 516   // g_buf row stride (f32): 4-bank lane skew, 16B-aligned rows
#define HST 136   // h_buf row stride (bf16): 4-bank lane skew, 16B-aligned rows

__global__ __launch_bounds__(256) void lstm_rec_kernel(const float* __restrict__ xg,
                                                       const __bf16* __restrict__ WHP,
                                                       __bf16* __restrict__ y_pack,
                                                       float* __restrict__ h_final) {
  __shared__ float  g_buf[16 * GST];
  __shared__ __bf16 h_buf[16 * HST];

  const int tid  = threadIdx.x;
  const int lane = tid & 31;
  const int wave = tid >> 5;          // 0..7
  const int hi   = lane >> 4;
  const int lm   = lane & 15;
  const int blk  = blockIdx.x;

  // --- Whh^T B-fragments from packed image -> registers (resident 512 steps) ---
  v16bf bw[4][4];
#pragma unroll
  for (int nt = 0; nt < 4; ++nt)
#pragma unroll
    for (int kt = 0; kt < 4; ++kt)
      bw[nt][kt] = *(const v16bf*)(WHP + ((((size_t)(wave * 4 + nt)) * 4 + kt) * 32 + lane) * 16);

  // zero h0
  for (int i = tid; i < 16 * HST; i += 256) h_buf[i] = (__bf16)0.0f;

  // elementwise ownership: thread -> (row em, 8 consecutive cols ejc*8..+7)
  const int em  = tid & 15;   // row varies across lanes -> LDS-bank friendly
  const int ejc = tid >> 4;   // 0..15
  // packed-A chunk coordinates for this thread's 8 outputs (j = ejc*8..+7)
  const int ykt    = ejc >> 2;                  // which 16x32 K-tile of layer-1 A
  const int ylane  = em + ((ejc & 1) << 4);     // lane in A image
  const int yhalf8 = ((ejc >> 1) & 1) * 8;      // element half offset
  float c[8];
#pragma unroll
  for (int q = 0; q < 8; ++q) c[q] = 0.0f;

  // software-pipelined xg fragments (tile-major, one v8f per tile per lane)
  v8f xgv[4];
#pragma unroll
  for (int nt = 0; nt < 4; ++nt)
    xgv[nt] = *(const v8f*)(xg + xg_off(blk, wave * 4 + nt, lane)); // rowtile(s=0)=blk

  __syncthreads();

  for (int s = 0; s < LS; ++s) {
    // A fragments (h rows of this block) from LDS — identical across waves
    v16bf af[4];
#pragma unroll
    for (int kt = 0; kt < 4; ++kt) {
#pragma unroll
      for (int e = 0; e < 16; ++e) {
        int k = kt * 32 + (e & 7) + ((e >> 3) << 4) + (hi << 3);
        af[kt][e] = h_buf[lm * HST + k];
      }
    }

#pragma unroll
    for (int nt = 0; nt < 4; ++nt) {
      const int n = wave * 64 + nt * 16 + lm;
      v8f acc = xgv[nt];                       // accum init = xg (incl. biases)
#pragma unroll
      for (int kt = 0; kt < 4; ++kt)
        acc = __builtin_amdgcn_wmma_f32_16x16x32_bf16(false, af[kt], false, bw[nt][kt],
                                                      (short)0, acc, false, false);
#pragma unroll
      for (int r = 0; r < 8; ++r)
        g_buf[(r + (hi << 3)) * GST + n] = acc[r];
    }

    // issue next step's xg loads now; they overlap the gate phase and the
    // s_wait_loadcnt lands at next iteration's accumulator init
    if (s + 1 < LS) {
      const int rt_next = (s + 1) * 16 + blk;
#pragma unroll
      for (int nt = 0; nt < 4; ++nt)
        xgv[nt] = *(const v8f*)(xg + xg_off(rt_next, wave * 4 + nt, lane));
      if (s + 2 < LS)
        __builtin_prefetch(xg + xg_off((s + 2) * 16 + blk, wave * 4, lane), 0, 1);
    }
    __syncthreads();

    // fused gates: torch order i,f,g,o in chunks of 128
    float hq[8];
#pragma unroll
    for (int q = 0; q < 8; ++q) {
      int j = ejc * 8 + q;
      float iv = sig_f (g_buf[em * GST + j]);
      float fv = sig_f (g_buf[em * GST + 128 + j]);
      float gv = tanh_f(g_buf[em * GST + 256 + j]);
      float ov = sig_f (g_buf[em * GST + 384 + j]);
      c[q] = fv * c[q] + iv * gv;
      hq[q] = ov * tanh_f(c[q]);
      h_buf[em * HST + j] = (__bf16)hq[q];
    }
    if (y_pack) {   // store directly as layer-1 A-fragment chunk (16B per thread)
      const size_t rowtile = (size_t)s * 16 + blk;
      v8bf t;
#pragma unroll
      for (int q = 0; q < 8; ++q) t[q] = (__bf16)hq[q];
      *(v8bf*)(y_pack + ((rowtile * 4 + ykt) * 32 + ylane) * 16 + yhalf8) = t;
    }
    if (h_final && s == LS - 1) {
#pragma unroll
      for (int q = 0; q < 8; ++q)
        h_final[(size_t)(blk * 16 + em) * LH + ejc * 8 + q] = hq[q];
    }
    __syncthreads();
  }
}

// ---------------------------------------------------------------------------
extern "C" void kernel_launch(void* const* d_in, const int* in_sizes, int n_in,
                              void* d_out, int out_size, void* d_ws, size_t ws_size,
                              hipStream_t stream) {
  const float* x    = (const float*)d_in[0];
  const float* Wih0 = (const float*)d_in[1];
  const float* Whh0 = (const float*)d_in[2];
  const float* bih0 = (const float*)d_in[3];
  const float* bhh0 = (const float*)d_in[4];
  const float* Wih1 = (const float*)d_in[5];
  const float* Whh1 = (const float*)d_in[6];
  const float* bih1 = (const float*)d_in[7];
  const float* bhh1 = (const float*)d_in[8];
  float* out = (float*)d_out;   // (B, H) f32

  const size_t ROWS = (size_t)LS * LB;                 // 131072
  char* ws = (char*)d_ws;
  size_t off = 0;
  __bf16* A0P  = (__bf16*)(ws + off); off += ROWS * LC * sizeof(__bf16);  // 16 MB
  float*  XG   = (float*) (ws + off); off += ROWS * LG * sizeof(float);   // 256 MB
  __bf16* Y1P  = (__bf16*)(ws + off); off += ROWS * LH * sizeof(__bf16);  // 32 MB
  __bf16* WP0  = (__bf16*)(ws + off); off += (size_t)LG * LC * sizeof(__bf16); // 64 KB
  __bf16* WP1  = (__bf16*)(ws + off); off += (size_t)LG * LH * sizeof(__bf16); // 128 KB
  __bf16* WHP0 = (__bf16*)(ws + off); off += (size_t)LG * LH * sizeof(__bf16); // 128 KB
  __bf16* WHP1 = (__bf16*)(ws + off); off += (size_t)LG * LH * sizeof(__bf16); // 128 KB

  // weight pre-pack (one-shot, tiny)
  pack_w_kernel<<<(NTILES * 2 * 32 + 255) / 256, 256, 0, stream>>>(Wih0, WP0, LC);
  pack_w_kernel<<<(NTILES * 4 * 32 + 255) / 256, 256, 0, stream>>>(Wih1, WP1, LH);
  pack_w_kernel<<<(NTILES * 4 * 32 + 255) / 256, 256, 0, stream>>>(Whh0, WHP0, LH);
  pack_w_kernel<<<(NTILES * 4 * 32 + 255) / 256, 256, 0, stream>>>(Whh1, WHP1, LH);
  // x -> packed layer-0 A fragments
  cvt_x_pack_kernel<<<(int)(ROWS * LC / 8 / 256), 256, 0, stream>>>(x, A0P);
  // layer-0 input GEMM (K=64): XG = A0 @ Wih0^T + bih0 + bhh0
  ingemm_kernel<LC><<<(int)(ROWS / 16), 256, 0, stream>>>(A0P, WP0, bih0, bhh0, XG);
  // layer-0 recurrence -> Y1 (packed layer-1 A fragments)
  lstm_rec_kernel<<<LB / 16, 256, 0, stream>>>(XG, WHP0, Y1P, nullptr);
  // layer-1 input GEMM (K=128): XG = Y1 @ Wih1^T + bih1 + bhh1
  ingemm_kernel<LH><<<(int)(ROWS / 16), 256, 0, stream>>>(Y1P, WP1, bih1, bhh1, XG);
  // layer-1 recurrence; last step writes f32 output
  lstm_rec_kernel<<<LB / 16, 256, 0, stream>>>(XG, WHP1, nullptr, out);
}
